// GCN_2018634629408
// MI455X (gfx1250) — matrix operand order repaired and checked by reading, not compiled, not run on hardware
//
#include <hip/hip_runtime.h>
#include <hip/hip_bf16.h>
#include <math.h>

// ---------------------------------------------------------------------------
// Types for WMMA (gfx1250, wave32)
// ---------------------------------------------------------------------------
typedef __attribute__((ext_vector_type(16))) _Float16 v16h;
typedef __attribute__((ext_vector_type(8)))  _Float16 v8h;
typedef __attribute__((ext_vector_type(8)))  float    v8f;

union V16U { v16h v; v8h h[2]; };

__device__ __forceinline__ v16h ld_frag2(const _Float16* lo, const _Float16* hi) {
    V16U u;
    u.h[0] = *(const v8h*)lo;   // 16B aligned (all row pitches are multiples of 8 halves)
    u.h[1] = *(const v8h*)hi;
    return u.v;
}

__device__ __forceinline__ v8f wmma_f16(v16h a, v16h b, v8f c) {
    return __builtin_amdgcn_wmma_f32_16x16x32_f16(
        /*neg_a=*/false, a, /*neg_b=*/false, b,
        /*c_mod=*/(short)0, c, /*reuse_a=*/false, /*reuse_b=*/false);
}

// ---------------------------------------------------------------------------
// Problem constants (from reference setup_inputs)
// ---------------------------------------------------------------------------
#define BB   64
#define NM   64
#define DM   6
#define FA   37
#define FB   6
#define NP   512
#define DP   12
#define FP   480
#define HM   128
#define HP   128

#define MROWS   (BB * NP)      // 32768 protein node rows (mult of 32)
#define K1PAD   480            // 15 * 32
#define N1PAD   224            // 14 * 16  (Wp1 cols 200 padded)
#define N2PAD   128            // 8 * 16   (Wp2 cols 100 padded)
#define N3PAD   128            // Wop cols exact

// ---------------------------------------------------------------------------
// Small utility kernels
// ---------------------------------------------------------------------------
__global__ void zero_f32(float* p, int n) {
    int i = blockIdx.x * blockDim.x + threadIdx.x;
    if (i < n) p[i] = 0.0f;
}

// Pack fp32 weight W[Korig,Norig] (row-major) into per-lane WMMA B fragments:
// layout: frag[ntile][ktile][lane][16 halves], lane = n + 16*g,
// half k' -> element (K = kt*32 + 8*g + k' + (k' >= 8 ? 8 : 0), N = nt*16 + n).  Zero-padded.
__global__ void pack_w(const float* __restrict__ W, _Float16* __restrict__ out,
                       int Korig, int Norig, int Kpad, int Npad) {
    int idx = blockIdx.x * blockDim.x + threadIdx.x;
    int KT = Kpad / 32;
    int total = (Npad / 16) * KT * 32 * 16;
    if (idx >= total) return;
    int kp   = idx & 15;
    int lane = (idx >> 4) & 31;
    int kt   = (idx >> 9) % KT;
    int nt   = (idx >> 9) / KT;
    int g = lane >> 4, n = lane & 15;
    int K = kt * 32 + g * 8 + kp + (kp >= 8 ? 8 : 0);
    int N = nt * 16 + n;
    float v = (K < Korig && N < Norig) ? W[(size_t)K * Norig + N] : 0.0f;
    out[idx] = (_Float16)v;
}

__global__ void pad_bias(const float* __restrict__ b, float* __restrict__ out,
                         int Norig, int Npad) {
    int i = blockIdx.x * blockDim.x + threadIdx.x;
    if (i < Npad) out[i] = (i < Norig) ? b[i] : 0.0f;
}

// ---------------------------------------------------------------------------
// Molecule branch (tiny -> scalar VALU kernels)
// ---------------------------------------------------------------------------
__global__ void bond_sum_k(const float* __restrict__ bonds, float* __restrict__ bsum) {
    int i = blockIdx.x * blockDim.x + threadIdx.x;      // over B*NM*FB
    if (i >= BB * NM * FB) return;
    int f = i % FB, node = i / FB;
    float s = 0.0f;
    for (int d = 0; d < DM; ++d) s += bonds[((size_t)node * DM + d) * FB + f];
    bsum[i] = s;
}

// relu( [X + nbr_sum(X), bond_sum] @ W + bias ) ; one block per node, 128 threads
__global__ void mol_gconv(const float* __restrict__ X, const float* __restrict__ bsum,
                          const int* __restrict__ edges, const float* __restrict__ W,
                          const float* __restrict__ bias, float* __restrict__ Y, int Fin) {
    __shared__ float xs[144];
    __shared__ int   e[DM];
    int node = blockIdx.x;          // B*NM
    int t = threadIdx.x;            // 128
    if (t < DM) e[t] = edges[node * DM + t];
    __syncthreads();
    int b = node >> 6;
    for (int f = t; f < Fin; f += 128) {
        float s = X[(size_t)node * Fin + f];
        for (int d = 0; d < DM; ++d)
            s += X[(size_t)(b * NM + e[d]) * Fin + f];
        xs[f] = s;
    }
    if (t < FB) xs[Fin + t] = bsum[node * FB + t];
    __syncthreads();
    int Kc = Fin + FB;
    float acc = bias[t];
    for (int k = 0; k < Kc; ++k) acc = fmaf(xs[k], W[(size_t)k * 128 + t], acc);
    Y[(size_t)node * 128 + t] = fmaxf(acc, 0.0f);
}

// fp_m[b,:] = sum_n tanh([A2, bond_sum] @ Wo + bo) ; one block per molecule
__global__ void mol_out(const float* __restrict__ A2, const float* __restrict__ bsum,
                        const float* __restrict__ Wo, const float* __restrict__ bo,
                        float* __restrict__ fpm) {
    __shared__ float xs[144];
    int b = blockIdx.x, t = threadIdx.x;   // 128 threads
    float acc = 0.0f;
    for (int n = 0; n < NM; ++n) {
        int node = b * NM + n;
        __syncthreads();
        xs[t] = A2[(size_t)node * 128 + t];
        if (t < FB) xs[128 + t] = bsum[node * FB + t];
        __syncthreads();
        float v = bo[t];
        for (int k = 0; k < 134; ++k) v = fmaf(xs[k], Wo[(size_t)k * 128 + t], v);
        acc += tanhf(v);
    }
    fpm[(size_t)b * 128 + t] = acc;
}

// ---------------------------------------------------------------------------
// Protein branch: gather-sum kernels (fp32 / f16 -> f16)
// ---------------------------------------------------------------------------
__global__ void pxsum1(const float* __restrict__ P, const int* __restrict__ edges,
                       _Float16* __restrict__ Xs) {
    __shared__ int e[DP];
    int i = blockIdx.x;            // B*NP
    int t = threadIdx.x;           // 256
    if (t < DP) e[t] = edges[i * DP + t];
    __syncthreads();
    int b = i >> 9;
    for (int f = t; f < FP; f += 256) {
        float s = P[(size_t)i * FP + f];
        for (int d = 0; d < DP; ++d)
            s += P[((size_t)(b * NP + e[d])) * FP + f];
        Xs[(size_t)i * FP + f] = (_Float16)s;
    }
}

__global__ void pxsum2(const _Float16* __restrict__ Y1, const int* __restrict__ edges,
                       _Float16* __restrict__ Xs) {
    __shared__ int e[DP];
    int i = blockIdx.x;
    int t = threadIdx.x;           // 256
    if (t < DP) e[t] = edges[i * DP + t];
    __syncthreads();
    int b = i >> 9;
    for (int f = t; f < N1PAD; f += 256) {
        float s = (float)Y1[(size_t)i * N1PAD + f];
        for (int d = 0; d < DP; ++d)
            s += (float)Y1[((size_t)(b * NP + e[d])) * N1PAD + f];
        Xs[(size_t)i * N1PAD + f] = (_Float16)s;
    }
}

// ---------------------------------------------------------------------------
// WMMA GEMM: Y = relu(A[M,Kpad] @ Bpack + bias), f16 in/out, f32 accumulate.
// Each wave computes a 32(M)x32(N) tile: MT=2 x NT=2 -> 4 accumulators,
// A and B fragments each reused 2x => 2 b128 loads per WMMA, and 4
// independent accumulation chains to cover WMMA->WMMA RAW latency.
// Grids are sized so every wave is fully active (EXEC all-ones for WMMA).
// ---------------------------------------------------------------------------
__global__ __launch_bounds__(256)
void gemm_relu_wmma(const _Float16* __restrict__ A, const _Float16* __restrict__ Bp,
                    const float* __restrict__ bias, _Float16* __restrict__ Y,
                    int M, int Kpad, int Npad) {
    const int KT = Kpad / 32, NG = Npad / 32;
    int w = (int)((blockIdx.x * blockDim.x + threadIdx.x) >> 5);
    int lane = threadIdx.x & 31;
    int total_w = (M / 32) * NG;
    if (w >= total_w) return;                 // wave-uniform (never fires for our grids)
    int mp = w / NG, ng = w % NG;
    int mt0 = mp * 2, nt0 = ng * 2;
    int g = lane >> 4, n = lane & 15;

    const _Float16* arow0 = A + (size_t)(mt0 * 16 + n) * Kpad;
    const _Float16* arow1 = arow0 + (size_t)16 * Kpad;
    v8f c00 = {}; v8f c01 = {}; v8f c10 = {}; v8f c11 = {};
    for (int kt = 0; kt < KT; ++kt) {
        int kb = kt * 32;
        v16h a0 = ld_frag2(arow0 + kb + 8 * g, arow0 + kb + 16 + 8 * g);
        v16h a1 = ld_frag2(arow1 + kb + 8 * g, arow1 + kb + 16 + 8 * g);
        const _Float16* b0p = Bp + ((size_t)(nt0 * KT + kt) * 32 + lane) * 16;
        const _Float16* b1p = Bp + ((size_t)((nt0 + 1) * KT + kt) * 32 + lane) * 16;
        v16h b0 = ld_frag2(b0p, b0p + 8);
        v16h b1 = ld_frag2(b1p, b1p + 8);
        c00 = wmma_f16(a0, b0, c00);
        c01 = wmma_f16(a0, b1, c01);
        c10 = wmma_f16(a1, b0, c10);
        c11 = wmma_f16(a1, b1, c11);
    }
    int N0 = nt0 * 16 + n, N1 = N0 + 16;
    float bb0 = bias[N0], bb1 = bias[N1];
    for (int r = 0; r < 8; ++r) {
        int row0 = mt0 * 16 + r + 8 * g;
        int row1 = row0 + 16;
        float v;
        v = c00[r] + bb0; Y[(size_t)row0 * Npad + N0] = (_Float16)fmaxf(v, 0.0f);
        v = c01[r] + bb1; Y[(size_t)row0 * Npad + N1] = (_Float16)fmaxf(v, 0.0f);
        v = c10[r] + bb0; Y[(size_t)row1 * Npad + N0] = (_Float16)fmaxf(v, 0.0f);
        v = c11[r] + bb1; Y[(size_t)row1 * Npad + N1] = (_Float16)fmaxf(v, 0.0f);
    }
}

// WMMA GEMM + tanh + per-molecule row reduction: fp[b,N] += sum_rows tanh(A@B + bias)
// 32x32 wave tile; rows_per_graph (512) is a multiple of 32 so a wave's 32 rows
// always belong to one molecule.
__global__ __launch_bounds__(256)
void gemm_tanh_rowsum_wmma(const _Float16* __restrict__ A, const _Float16* __restrict__ Bp,
                           const float* __restrict__ bias, float* __restrict__ fpout,
                           int M, int Kpad, int Npad, int rows_per_graph) {
    const int KT = Kpad / 32, NG = Npad / 32;
    int w = (int)((blockIdx.x * blockDim.x + threadIdx.x) >> 5);
    int lane = threadIdx.x & 31;
    int total_w = (M / 32) * NG;
    if (w >= total_w) return;
    int mp = w / NG, ng = w % NG;
    int mt0 = mp * 2, nt0 = ng * 2;
    int g = lane >> 4, n = lane & 15;

    const _Float16* arow0 = A + (size_t)(mt0 * 16 + n) * Kpad;
    const _Float16* arow1 = arow0 + (size_t)16 * Kpad;
    v8f c00 = {}; v8f c01 = {}; v8f c10 = {}; v8f c11 = {};
    for (int kt = 0; kt < KT; ++kt) {
        int kb = kt * 32;
        v16h a0 = ld_frag2(arow0 + kb + 8 * g, arow0 + kb + 16 + 8 * g);
        v16h a1 = ld_frag2(arow1 + kb + 8 * g, arow1 + kb + 16 + 8 * g);
        const _Float16* b0p = Bp + ((size_t)(nt0 * KT + kt) * 32 + lane) * 16;
        const _Float16* b1p = Bp + ((size_t)((nt0 + 1) * KT + kt) * 32 + lane) * 16;
        v16h b0 = ld_frag2(b0p, b0p + 8);
        v16h b1 = ld_frag2(b1p, b1p + 8);
        c00 = wmma_f16(a0, b0, c00);
        c01 = wmma_f16(a0, b1, c01);
        c10 = wmma_f16(a1, b0, c10);
        c11 = wmma_f16(a1, b1, c11);
    }
    int N0 = nt0 * 16 + n, N1 = N0 + 16;
    float bb0 = bias[N0], bb1 = bias[N1];
    float s0 = 0.0f, s1 = 0.0f;
    for (int r = 0; r < 8; ++r) {
        s0 += tanhf(c00[r] + bb0) + tanhf(c10[r] + bb0);
        s1 += tanhf(c01[r] + bb1) + tanhf(c11[r] + bb1);
    }
    // lane n and lane n+16 hold the two row-halves of column n
    s0 += __shfl_xor(s0, 16, 32);
    s1 += __shfl_xor(s1, 16, 32);
    int b = (mt0 * 16) / rows_per_graph;
    if (g == 0) {
        atomicAdd(&fpout[(size_t)b * Npad + N0], s0);
        atomicAdd(&fpout[(size_t)b * Npad + N1], s1);
    }
}

// ---------------------------------------------------------------------------
// FC head: fp=[fp_m,fp_p]; h2 = (fp@Wf1+bf1)@Wf2+bf2; out = sigmoid(h2@Wout+bout)
// ---------------------------------------------------------------------------
__global__ void fc_head(const float* __restrict__ fpm, const float* __restrict__ fpp,
                        const float* __restrict__ Wf1, const float* __restrict__ bf1,
                        const float* __restrict__ Wf2, const float* __restrict__ bf2,
                        const float* __restrict__ Wout, const float* __restrict__ bout,
                        float* __restrict__ out) {
    __shared__ float fp[256], h1[100], h2[100];
    int b = blockIdx.x, t = threadIdx.x;   // 128 threads
    fp[t]       = fpm[(size_t)b * 128 + t];
    fp[128 + t] = fpp[(size_t)b * 128 + t];
    __syncthreads();
    if (t < 100) {
        float a = bf1[t];
        for (int k = 0; k < 256; ++k) a = fmaf(fp[k], Wf1[(size_t)k * 100 + t], a);
        h1[t] = a;
    }
    __syncthreads();
    if (t < 100) {
        float a = bf2[t];
        for (int k = 0; k < 100; ++k) a = fmaf(h1[k], Wf2[(size_t)k * 100 + t], a);
        h2[t] = a;
    }
    __syncthreads();
    if (t == 0) {
        float o = bout[0];
        for (int k = 0; k < 100; ++k) o = fmaf(h2[k], Wout[k], o);
        out[b] = 1.0f / (1.0f + expf(-o));
    }
}

// ---------------------------------------------------------------------------
// Launcher
// ---------------------------------------------------------------------------
static inline size_t alignup(size_t x) { return (x + 255) & ~(size_t)255; }

extern "C" void kernel_launch(void* const* d_in, const int* in_sizes, int n_in,
                              void* d_out, int out_size, void* d_ws, size_t ws_size,
                              hipStream_t stream) {
    const float* m_atoms = (const float*)d_in[0];
    const float* m_bonds = (const float*)d_in[1];
    const int*   m_edges = (const int*)  d_in[2];
    const float* p_atoms = (const float*)d_in[3];
    const int*   p_edges = (const int*)  d_in[4];
    const float* W1  = (const float*)d_in[5];   const float* b1  = (const float*)d_in[6];
    const float* W2  = (const float*)d_in[7];   const float* b2  = (const float*)d_in[8];
    const float* Wp1 = (const float*)d_in[9];   const float* bp1 = (const float*)d_in[10];
    const float* Wp2 = (const float*)d_in[11];  const float* bp2 = (const float*)d_in[12];
    const float* Wo  = (const float*)d_in[13];  const float* bo  = (const float*)d_in[14];
    const float* Wop = (const float*)d_in[15];  const float* bop = (const float*)d_in[16];
    const float* Wf1 = (const float*)d_in[17];  const float* bf1 = (const float*)d_in[18];
    const float* Wf2 = (const float*)d_in[19];  const float* bf2 = (const float*)d_in[20];
    const float* Wout= (const float*)d_in[21];  const float* bout= (const float*)d_in[22];
    float* out = (float*)d_out;

    // ---- workspace carve-out -------------------------------------------------
    char* base = (char*)d_ws;
    size_t off = 0;
    auto carve = [&](size_t bytes) { void* p = base + off; off += alignup(bytes); return p; };

    float*    ws_bsum = (float*)   carve((size_t)BB * NM * FB * 4);
    float*    ws_a1   = (float*)   carve((size_t)BB * NM * 128 * 4);
    float*    ws_a2   = (float*)   carve((size_t)BB * NM * 128 * 4);
    float*    ws_fpm  = (float*)   carve((size_t)BB * HM * 4);
    float*    ws_fpp  = (float*)   carve((size_t)BB * HP * 4);
    _Float16* ws_Xs1  = (_Float16*)carve((size_t)MROWS * K1PAD * 2);
    _Float16* ws_Y1   = (_Float16*)carve((size_t)MROWS * N1PAD * 2);
    _Float16* ws_Xs2  = (_Float16*)carve((size_t)MROWS * N1PAD * 2);
    _Float16* ws_Y2   = (_Float16*)carve((size_t)MROWS * N2PAD * 2);
    _Float16* ws_Bp1  = (_Float16*)carve((size_t)(N1PAD/16) * (K1PAD/32) * 512 * 2);
    _Float16* ws_Bp2  = (_Float16*)carve((size_t)(N2PAD/16) * (N1PAD/32) * 512 * 2);
    _Float16* ws_Bp3  = (_Float16*)carve((size_t)(N3PAD/16) * (N2PAD/32) * 512 * 2);
    float*    ws_bia1 = (float*)   carve((size_t)N1PAD * 4);
    float*    ws_bia2 = (float*)   carve((size_t)N2PAD * 4);
    float*    ws_bia3 = (float*)   carve((size_t)N3PAD * 4);
    (void)ws_size; (void)in_sizes; (void)n_in; (void)out_size;

    // ---- weight / bias packing ----------------------------------------------
    {
        int tot1 = (N1PAD/16) * (K1PAD/32) * 512;   // 107520
        int tot2 = (N2PAD/16) * (N1PAD/32) * 512;   // 28672
        int tot3 = (N3PAD/16) * (N2PAD/32) * 512;   // 16384
        pack_w<<<(tot1 + 255) / 256, 256, 0, stream>>>(Wp1, ws_Bp1, 480, 200, K1PAD, N1PAD);
        pack_w<<<(tot2 + 255) / 256, 256, 0, stream>>>(Wp2, ws_Bp2, 200, 100, N1PAD, N2PAD);
        pack_w<<<(tot3 + 255) / 256, 256, 0, stream>>>(Wop, ws_Bp3, 100, 128, N2PAD, N3PAD);
        pad_bias<<<1, 256, 0, stream>>>(bp1, ws_bia1, 200, N1PAD);
        pad_bias<<<1, 128, 0, stream>>>(bp2, ws_bia2, 100, N2PAD);
        pad_bias<<<1, 128, 0, stream>>>(bop, ws_bia3, 128, N3PAD);
        zero_f32<<<(BB * HP + 255) / 256, 256, 0, stream>>>(ws_fpp, BB * HP);
    }

    // ---- molecule branch (tiny, scalar) -------------------------------------
    bond_sum_k<<<(BB * NM * FB + 255) / 256, 256, 0, stream>>>(m_bonds, ws_bsum);
    mol_gconv<<<BB * NM, 128, 0, stream>>>(m_atoms, ws_bsum, m_edges, W1, b1, ws_a1, FA);
    mol_gconv<<<BB * NM, 128, 0, stream>>>(ws_a1,   ws_bsum, m_edges, W2, b2, ws_a2, 128);
    mol_out  <<<BB, 128, 0, stream>>>(ws_a2, ws_bsum, Wo, bo, ws_fpm);

    // ---- protein branch: gather-sums + WMMA GEMMs ---------------------------
    pxsum1<<<MROWS, 256, 0, stream>>>(p_atoms, p_edges, ws_Xs1);
    {   // GEMM1: [32768,480] x [480,224] -> relu -> Y1   (32x32 tiles per wave)
        int waves = (MROWS / 32) * (N1PAD / 32);            // 1024*7 = 7168
        gemm_relu_wmma<<<waves * 32 / 256, 256, 0, stream>>>(
            ws_Xs1, ws_Bp1, ws_bia1, ws_Y1, MROWS, K1PAD, N1PAD);
    }
    pxsum2<<<MROWS, 256, 0, stream>>>(ws_Y1, p_edges, ws_Xs2);
    {   // GEMM2: [32768,224] x [224,128] -> relu -> Y2
        int waves = (MROWS / 32) * (N2PAD / 32);            // 1024*4 = 4096
        gemm_relu_wmma<<<waves * 32 / 256, 256, 0, stream>>>(
            ws_Xs2, ws_Bp2, ws_bia2, ws_Y2, MROWS, N1PAD, N2PAD);
    }
    {   // GEMM3: [32768,128] x [128,128] -> tanh -> per-molecule row sum -> fp_p
        int waves = (MROWS / 32) * (N3PAD / 32);            // 4096
        gemm_tanh_rowsum_wmma<<<waves * 32 / 256, 256, 0, stream>>>(
            ws_Y2, ws_Bp3, ws_bia3, ws_fpp, MROWS, N2PAD, N3PAD, NP);
    }

    // ---- FC head + sigmoid ---------------------------------------------------
    fc_head<<<BB, 128, 0, stream>>>(ws_fpm, ws_fpp, Wf1, bf1, Wf2, bf2, Wout, bout, out);
}
